// TransformerSLN_90211493085601
// MI455X (gfx1250) — matrix operand order repaired and checked by reading, not compiled, not run on hardware
//
#include <hip/hip_runtime.h>
#include <math.h>

// ---------------- problem constants ----------------
constexpr int CB   = 8;            // batch
constexpr int CS   = 1024;         // sequence
constexpr int CD   = 768;          // model dim
constexpr int CH   = 8;            // heads
constexpr int CDH  = 96;           // head dim
constexpr int CDFF = 3072;         // mlp hidden
constexpr int CM   = CB * CS;      // 8192 rows
constexpr float CEPS = 1e-5f;

// ---------------- helpers ----------------
typedef __attribute__((ext_vector_type(16))) __bf16 v16bf;
typedef __attribute__((ext_vector_type(8)))  float  v8f;

union Frag { v16bf v; uint4 q[2]; };

__device__ __forceinline__ unsigned short f2bf(float f) {
    unsigned u = __float_as_uint(f);
    u += 0x7FFFu + ((u >> 16) & 1u);      // round-to-nearest-even
    return (unsigned short)(u >> 16);
}
__device__ __forceinline__ float b2f(unsigned short u) {
    return __uint_as_float((unsigned)u << 16);
}

// ---------------- bf16 WMMA GEMM ----------------
// 128x128 block tile, 8 waves x 32x64, LDS double buffer + register prefetch.
// A: bf16 [M][K] row-major (lda). Dims assumed: M % 128 == 0, K % 32 == 0.
// BT=true : B bf16 [N][K] row-major (ldb), N % 128 == 0 -> unguarded staging.
// BT=false: B bf16 [K][N] row-major (ldb), transpose-staged with N guards.
// Output: bf16 to Cb if non-null else f32 to Cf (ntC=1 -> non-temporal stores).
// Batch offset for X: (z/batH)*sXb + (z%batH)*sXh (element counts).
template<bool BT>
__global__ __launch_bounds__(256)
void gemm_bf16_wmma(const unsigned short* __restrict__ A,
                    const unsigned short* __restrict__ Bm,
                    const float* __restrict__ bias, const float* __restrict__ Res,
                    float* __restrict__ Cf, unsigned short* __restrict__ Cb,
                    int M, int N, int K, int lda, int ldb, int ldc,
                    long long sAb, long long sAh, long long sBb, long long sBh,
                    long long sCb, long long sCh, int batH, int doRelu, int ntC)
{
    __shared__ __align__(16) unsigned short As[2][128][32];   // 16KB
    __shared__ __align__(16) unsigned short Bs[2][128][32];   // 16KB

    const int tid  = threadIdx.x;
    const int lane = tid & 31;
    const int wave = tid >> 5;
    const int wm   = wave & 3;        // 4 waves along M, 32 rows each
    const int wn   = wave >> 2;       // 2 waves along N, 64 cols each

    const int z     = blockIdx.z;
    const int tileM = blockIdx.y * 128;
    const int tileN = blockIdx.x * 128;

    const long long offA = (long long)(z / batH) * sAb + (long long)(z % batH) * sAh;
    const long long offB = (long long)(z / batH) * sBb + (long long)(z % batH) * sBh;
    const long long offC = (long long)(z / batH) * sCb + (long long)(z % batH) * sCh;
    const unsigned short* Ab = A  + offA;
    const unsigned short* Bb = Bm + offB;

    v8f acc[2][4];
#pragma unroll
    for (int i = 0; i < 2; ++i)
#pragma unroll
        for (int j = 0; j < 4; ++j) acc[i][j] = (v8f){};

    const int am0 = wm * 32 + (lane & 15);       // A frag rows
    const int am1 = am0 + 16;
    const int kh  = (lane >> 4) * 8;             // A: runs at kh, kh+16
    const int kb  = (lane >> 4) * 8;             // B: runs at kb, kb+16
    const int bnB = wn * 64 + (lane & 15);       // B frag cols base

    // per-thread staging coordinates (uint2 = 4 bf16 chunks, 4 per thread)
    const int sr  = tid >> 3;                    // row 0..31 step: +32 per i
    const int sc4 = (tid & 7) * 4;               // k offset 0..28

    // ---- combined staging (prologue + NT path) ----
    auto stage = [&](int buf, int kt) {
#pragma unroll
        for (int i = 0; i < 4; ++i) {
            const int r = sr + i * 32;
            *(uint2*)&As[buf][r][sc4] =
                *(const uint2*)(Ab + (long long)(tileM + r) * lda + (kt + sc4));
        }
        if (BT) {
#pragma unroll
            for (int i = 0; i < 4; ++i) {
                const int r = sr + i * 32;
                *(uint2*)&Bs[buf][r][sc4] =
                    *(const uint2*)(Bb + (long long)(tileN + r) * ldb + (kt + sc4));
            }
        } else {
            // transpose-stage [K][N] -> Bs[n][k], guarded on N (only attn@V, N=96)
#pragma unroll
            for (int i = 0; i < 4; ++i) {
                const int idx = tid + i * 256;
                const int kk = idx >> 5;         // 0..31
                const int n4 = (idx & 31) * 4;   // 0..124
                const unsigned short* src = Bb + (long long)(kt + kk) * ldb;
#pragma unroll
                for (int j = 0; j < 4; ++j) {
                    const int gn = tileN + n4 + j;
                    Bs[buf][n4 + j][kk] = (gn < N) ? src[gn] : (unsigned short)0;
                }
            }
        }
    };

    const int nk = K >> 5;            // K / 32 steps
    stage(0, 0);
    __syncthreads();

    for (int it = 0; it < nk; ++it) {
        const int cur = it & 1;
        // branchless clamped next tile (last iter redundantly reloads last tile)
        const int ktn = (it + 1 < nk) ? (it + 1) * 32 : it * 32;

        uint2 ra[4], rb[4];
        if (BT) {
            // issue next-tile global loads first; hide latency under the WMMAs
#pragma unroll
            for (int i = 0; i < 4; ++i) {
                const int r = sr + i * 32;
                ra[i] = *(const uint2*)(Ab + (long long)(tileM + r) * lda + (ktn + sc4));
                rb[i] = *(const uint2*)(Bb + (long long)(tileN + r) * ldb + (ktn + sc4));
            }
        } else {
            stage(cur ^ 1, ktn);
        }

        Frag fa0, fa1;
        fa0.q[0] = *(const uint4*)&As[cur][am0][kh];
        fa0.q[1] = *(const uint4*)&As[cur][am0][kh + 16];
        fa1.q[0] = *(const uint4*)&As[cur][am1][kh];
        fa1.q[1] = *(const uint4*)&As[cur][am1][kh + 16];
#pragma unroll
        for (int f = 0; f < 4; ++f) {
            Frag fb;
            fb.q[0] = *(const uint4*)&Bs[cur][bnB + f * 16][kb];
            fb.q[1] = *(const uint4*)&Bs[cur][bnB + f * 16][kb + 16];
            acc[0][f] = __builtin_amdgcn_wmma_f32_16x16x32_bf16(false, fa0.v, false, fb.v,
                                                                (short)0, acc[0][f], false, false);
            acc[1][f] = __builtin_amdgcn_wmma_f32_16x16x32_bf16(false, fa1.v, false, fb.v,
                                                                (short)0, acc[1][f], false, false);
        }

        if (BT) {
            // park next tile into the alternate LDS buffer
#pragma unroll
            for (int i = 0; i < 4; ++i) {
                const int r = sr + i * 32;
                *(uint2*)&As[cur ^ 1][r][sc4] = ra[i];
                *(uint2*)&Bs[cur ^ 1][r][sc4] = rb[i];
            }
        }
        __syncthreads();   // single barrier: next-buffer stores + this-buffer reads
    }

    // ---- epilogue: lane holds col (lane&15), rows r + 8*(lane>>4) ----
#pragma unroll
    for (int mi = 0; mi < 2; ++mi) {
        const int rowBase = tileM + wm * 32 + mi * 16 + (lane >> 4) * 8;
#pragma unroll
        for (int f = 0; f < 4; ++f) {
            const int col = tileN + wn * 64 + f * 16 + (lane & 15);
            if (col >= N) continue;
            const float bv = bias ? bias[col] : 0.f;
#pragma unroll
            for (int r = 0; r < 8; ++r) {
                const long long ci = offC + (long long)(rowBase + r) * ldc + col;
                float val = acc[mi][f][r] + bv;
                if (doRelu) val = fmaxf(val, 0.f);
                if (Res) val += Res[ci];
                if (Cb)      Cb[ci] = f2bf(val);
                else if (ntC) __builtin_nontemporal_store(val, &Cf[ci]);
                else         Cf[ci] = val;
            }
        }
    }
}

// ---------------- weight transpose+convert: W f32 [K][N] -> Wt bf16 [N][K] ----------------
__global__ __launch_bounds__(256)
void transpose_w_bf16(const float* __restrict__ W, unsigned short* __restrict__ Wt,
                      int K, int N)
{
    __shared__ float t[32][33];
    const int kb = blockIdx.y * 32, nb = blockIdx.x * 32;
    const int tx = threadIdx.x & 31, ty = threadIdx.x >> 5;   // ty: 0..7
#pragma unroll
    for (int i = ty; i < 32; i += 8)
        t[i][tx] = W[(long long)(kb + i) * N + (nb + tx)];
    __syncthreads();
#pragma unroll
    for (int i = ty; i < 32; i += 8)
        Wt[(long long)(nb + i) * K + (kb + tx)] = f2bf(t[tx][i]);
}

// ---------------- SLN: out_bf16 = gamma*x*LayerNorm(h) + beta*x ----------------
__global__ __launch_bounds__(256)
void sln_kernel(const float* __restrict__ h, const float* __restrict__ x,
                const float* __restrict__ g, const float* __restrict__ be,
                const float* __restrict__ w, const float* __restrict__ b,
                unsigned short* __restrict__ outb)
{
    __shared__ float s1[256], s2[256];
    const int row = blockIdx.x;
    const int tid = threadIdx.x;
    const long long base = (long long)row * CD;

    float sum = 0.f, sq = 0.f;
    for (int i = tid; i < CD; i += 256) {
        float v = h[base + i];
        sum += v; sq += v * v;
    }
    s1[tid] = sum; s2[tid] = sq;
    __syncthreads();
    for (int o = 128; o > 0; o >>= 1) {
        if (tid < o) { s1[tid] += s1[tid + o]; s2[tid] += s2[tid + o]; }
        __syncthreads();
    }
    const float mu  = s1[0] / (float)CD;
    const float var = s2[0] / (float)CD - mu * mu;
    const float rs  = rsqrtf(var + CEPS);
    const float g0 = g[0], be0 = be[0];

    for (int i = tid; i < CD; i += 256) {
        float ln = (h[base + i] - mu) * rs * w[i] + b[i];
        float xv = x[base + i];
        outb[base + i] = f2bf(g0 * xv * ln + be0 * xv);
    }
}

// ---------------- per-head row sum of squares from bf16 k ----------------
__global__ __launch_bounds__(256)
void rowsumsq_heads(const unsigned short* __restrict__ kbuf, float* __restrict__ k2)
{
    const int idx = blockIdx.x * 256 + threadIdx.x;     // over B*H*S
    if (idx >= CB * CH * CS) return;
    const int s  = idx % CS;
    const int hh = (idx / CS) % CH;
    const int b  = idx / (CS * CH);
    const unsigned short* row = kbuf + ((long long)(b * CS + s)) * CD + hh * CDH;
    float acc = 0.f;
    for (int j = 0; j < CDH; ++j) { float t = b2f(row[j]); acc += t * t; }
    k2[idx] = acc;
}

// ---------------- L2 softmax: read f32 scores (NT), write bf16 probs in-place ------------
// probs row s of batch z lives in the first 2KB of the 4KB f32 score row.
__global__ __launch_bounds__(256)
void attn_softmax(const unsigned short* __restrict__ qbuf, const float* __restrict__ k2,
                  float* __restrict__ scores)
{
    __shared__ float red[256];
    const int s = blockIdx.x;        // query row
    const int z = blockIdx.y;        // b*H + h
    const int b = z >> 3, hh = z & 7;
    const int tid = threadIdx.x;

    const unsigned short* qrow = qbuf + ((long long)(b * CS + s)) * CD + hh * CDH;
    float acc = 0.f;
    for (int j = tid; j < CDH; j += 256) { float t = b2f(qrow[j]); acc += t * t; }
    red[tid] = acc; __syncthreads();
    for (int o = 128; o > 0; o >>= 1) { if (tid < o) red[tid] += red[tid + o]; __syncthreads(); }
    const float q2v = red[0];
    __syncthreads();

    float* srow = scores + (long long)z * CS * CS + (long long)s * CS;
    const float* k2row = k2 + (long long)z * CS;
    const float scale = rsqrtf((float)CDH);

    float vals[CS / 256];
    float lmax = -1e30f;
#pragma unroll
    for (int t = 0; t < CS / 256; ++t) {
        const int col = tid + t * 256;
        float qk = __builtin_nontemporal_load(&srow[col]);   // streamed, don't pollute L2
        float d2 = q2v + k2row[col] - 2.0f * qk;
        d2 = d2 > 0.f ? d2 : 0.f;
        float lg = -sqrtf(d2) * scale;
        vals[t] = lg;
        lmax = fmaxf(lmax, lg);
    }
    red[tid] = lmax; __syncthreads();
    for (int o = 128; o > 0; o >>= 1) { if (tid < o) red[tid] = fmaxf(red[tid], red[tid + o]); __syncthreads(); }
    const float m = red[0]; __syncthreads();

    float sum = 0.f;
#pragma unroll
    for (int t = 0; t < CS / 256; ++t) { vals[t] = __expf(vals[t] - m); sum += vals[t]; }
    red[tid] = sum; __syncthreads();
    for (int o = 128; o > 0; o >>= 1) { if (tid < o) red[tid] += red[tid + o]; __syncthreads(); }
    const float inv = 1.0f / red[0];

    // bf16 probs packed into the front half of this row's f32 storage
    unsigned short* prow = (unsigned short*)srow;     // row-local, safe: reads done above
    __syncthreads();
#pragma unroll
    for (int t = 0; t < CS / 256; ++t) prow[tid + t * 256] = f2bf(vals[t] * inv);
}

// ---------------- plain copy ----------------
__global__ __launch_bounds__(256)
void copy_f32(const float* __restrict__ src, float* __restrict__ dst, long long n)
{
    long long i = (long long)blockIdx.x * 256 + threadIdx.x;
    if (i < n) dst[i] = src[i];
}

// ---------------- driver ----------------
extern "C" void kernel_launch(void* const* d_in, const int* in_sizes, int n_in,
                              void* d_out, int out_size, void* d_ws, size_t ws_size,
                              hipStream_t stream)
{
    const float* h    = (const float*)d_in[0];
    const float* x    = (const float*)d_in[1];
    const float* g1   = (const float*)d_in[2];
    const float* be1  = (const float*)d_in[3];
    const float* ln1w = (const float*)d_in[4];
    const float* ln1b = (const float*)d_in[5];
    const float* g2   = (const float*)d_in[6];
    const float* be2  = (const float*)d_in[7];
    const float* ln2w = (const float*)d_in[8];
    const float* ln2b = (const float*)d_in[9];
    const float* Wq = (const float*)d_in[10]; const float* bq = (const float*)d_in[11];
    const float* Wk = (const float*)d_in[12]; const float* bk = (const float*)d_in[13];
    const float* Wv = (const float*)d_in[14]; const float* bv = (const float*)d_in[15];
    const float* Wo = (const float*)d_in[16]; const float* bo = (const float*)d_in[17];
    const float* W1 = (const float*)d_in[18]; const float* b1 = (const float*)d_in[19];
    const float* W2 = (const float*)d_in[20]; const float* b2 = (const float*)d_in[21];

    float* out = (float*)d_out;

    const long long nMD  = (long long)CM * CD;          // 6,291,456
    const long long nMF  = (long long)CM * CDFF;
    const long long nW   = (long long)CD * CD;
    const long long nW1  = (long long)CD * CDFF;

    // workspace carve-up
    char* p = (char*)d_ws;
    auto takeF = [&](long long n) { float* r = (float*)p; p += n * sizeof(float); return r; };
    auto takeB = [&](long long n) { unsigned short* r = (unsigned short*)p; p += n * sizeof(unsigned short);
                                    p = (char*)(((uintptr_t)p + 15) & ~(uintptr_t)15); return r; };

    unsigned short* h1b   = takeB(nMD);
    unsigned short* qb    = takeB(nMD);
    unsigned short* kb    = takeB(nMD);
    unsigned short* vb    = takeB(nMD);
    unsigned short* attnb = takeB(nMD);
    unsigned short* h2b   = takeB(nMD);
    unsigned short* hidb  = takeB(nMF);
    unsigned short* Wqt   = takeB(nW);
    unsigned short* Wkt   = takeB(nW);
    unsigned short* Wvt   = takeB(nW);
    unsigned short* Wot   = takeB(nW);
    unsigned short* W1t   = takeB(nW1);
    unsigned short* W2t   = takeB(nW1);
    float* htmp   = takeF(nMD);
    float* k2buf  = takeF((long long)CB * CH * CS);
    float* scores = takeF((long long)CB * CH * CS * CS);   // 256 MB

    const dim3 blk(256);

    // out[0] = x
    copy_f32<<<dim3((unsigned)((nMD + 255) / 256)), blk, 0, stream>>>(x, out, nMD);

    // weights -> bf16 [N][K]
    transpose_w_bf16<<<dim3(CD / 32, CD / 32), blk, 0, stream>>>(Wq, Wqt, CD, CD);
    transpose_w_bf16<<<dim3(CD / 32, CD / 32), blk, 0, stream>>>(Wk, Wkt, CD, CD);
    transpose_w_bf16<<<dim3(CD / 32, CD / 32), blk, 0, stream>>>(Wv, Wvt, CD, CD);
    transpose_w_bf16<<<dim3(CD / 32, CD / 32), blk, 0, stream>>>(Wo, Wot, CD, CD);
    transpose_w_bf16<<<dim3(CDFF / 32, CD / 32), blk, 0, stream>>>(W1, W1t, CD, CDFF);
    transpose_w_bf16<<<dim3(CD / 32, CDFF / 32), blk, 0, stream>>>(W2, W2t, CDFF, CD);

    // h1 = SLN(h, x) -> bf16
    sln_kernel<<<dim3(CM), blk, 0, stream>>>(h, x, g1, be1, ln1w, ln1b, h1b);

    // q/k/v = h1 @ W* + b*  (bf16 out)
    const dim3 gProj(CD / 128, CM / 128, 1);
    gemm_bf16_wmma<true><<<gProj, blk, 0, stream>>>(h1b, Wqt, bq, nullptr, nullptr, qb,
        CM, CD, CD, CD, CD, CD, 0, 0, 0, 0, 0, 0, 1, 0, 0);
    gemm_bf16_wmma<true><<<gProj, blk, 0, stream>>>(h1b, Wkt, bk, nullptr, nullptr, kb,
        CM, CD, CD, CD, CD, CD, 0, 0, 0, 0, 0, 0, 1, 0, 0);
    gemm_bf16_wmma<true><<<gProj, blk, 0, stream>>>(h1b, Wvt, bv, nullptr, nullptr, vb,
        CM, CD, CD, CD, CD, CD, 0, 0, 0, 0, 0, 0, 1, 0, 0);

    // k2 row sums of squares
    rowsumsq_heads<<<dim3((CB * CH * CS + 255) / 256), blk, 0, stream>>>(kb, k2buf);

    // scores = q_head @ k_head^T (f32 NT out), batched over b,h
    const long long sQb = (long long)CS * CD, sQh = CDH;
    const long long sScF = (long long)CS * CS;        // f32 elements per z
    const dim3 gScore(CS / 128, CS / 128, CB * CH);
    gemm_bf16_wmma<true><<<gScore, blk, 0, stream>>>(qb, kb, nullptr, nullptr, scores, nullptr,
        CS, CS, CDH, CD, CD, CS,
        sQb, sQh, sQb, sQh, sScF * CH, sScF, CH, 0, 1);

    // softmax(-cdist/sqrt(DH)); writes bf16 probs in place (row stride 2048 shorts)
    attn_softmax<<<dim3(CS, CB * CH), blk, 0, stream>>>(qb, k2buf, scores);

    // attn_out = probs @ v_head (bf16 out); A = bf16 probs inside score buffer
    const long long sPr = 2ll * CS * CS;              // shorts per z
    const dim3 gAV((CDH + 127) / 128, CS / 128, CB * CH);
    gemm_bf16_wmma<false><<<gAV, blk, 0, stream>>>((const unsigned short*)scores, vb,
        nullptr, nullptr, nullptr, attnb,
        CS, CDH, CS, 2 * CS, CD, CD,
        sPr * CH, sPr, sQb, sQh, sQb, sQh, CH, 0, 0);

    // htmp = attn_out @ Wo + bo + h   (f32 out)
    gemm_bf16_wmma<true><<<gProj, blk, 0, stream>>>(attnb, Wot, bo, h, htmp, nullptr,
        CM, CD, CD, CD, CD, CD, 0, 0, 0, 0, 0, 0, 1, 0, 0);

    // h2 = SLN(htmp, x) -> bf16
    sln_kernel<<<dim3(CM), blk, 0, stream>>>(htmp, x, g2, be2, ln2w, ln2b, h2b);

    // hid = relu(h2 @ W1 + b1) (bf16 out)
    const dim3 gFF1(CDFF / 128, CM / 128, 1);
    gemm_bf16_wmma<true><<<gFF1, blk, 0, stream>>>(h2b, W1t, b1, nullptr, nullptr, hidb,
        CM, CDFF, CD, CD, CD, CDFF, 0, 0, 0, 0, 0, 0, 1, 1, 0);

    // out[1] = hid @ W2 + b2 + htmp  (f32 out)
    gemm_bf16_wmma<true><<<gProj, blk, 0, stream>>>(hidb, W2t, b2, htmp, out + nMD, nullptr,
        CM, CD, CDFF, CDFF, CDFF, CD, 0, 0, 0, 0, 0, 0, 1, 0, 0);
}
